// GNNGeneric_4148938408418
// MI455X (gfx1250) — matrix operand order repaired and checked by reading, not compiled, not run on hardware
//
#include <hip/hip_runtime.h>
#include <math.h>

// ---------------------------------------------------------------------------
// GNN SuperEdgeConv x3 + fusion MLP for MI455X (gfx1250, wave32, WMMA).
// Edge MLPs (~130 GFLOP hot spot) are wave-local bf16 WMMA GEMMs
// (v_wmma_f32_16x16x32_bf16) with features built directly in LDS, never
// materialized in HBM. Each wave computes a 32-row (MT=2) tile so every
// pre-packed B fragment (1KB/wave from L2) feeds two WMMAs, halving L2
// weight traffic vs a 16-row tile. segment_max is a native int-bit atomicMax
// (valid since values are post-ReLU >= 0 and empty segments must read 0).
// ---------------------------------------------------------------------------

typedef __bf16 bf16_t;
typedef bf16_t v16bf __attribute__((ext_vector_type(16)));
typedef bf16_t v8bf  __attribute__((ext_vector_type(8)));
typedef float  v8f   __attribute__((ext_vector_type(8)));

__device__ __forceinline__ bf16_t f2bf(float f) {
    unsigned u = __float_as_uint(f);
    unsigned r = u + 0x7FFFu + ((u >> 16) & 1u);   // round-to-nearest-even
    unsigned short h = (unsigned short)(r >> 16);
    return __builtin_bit_cast(bf16_t, h);
}

template<int NACC>
__device__ __forceinline__ void zero_acc(v8f* acc) {
#pragma unroll
    for (int t = 0; t < NACC; ++t)
#pragma unroll
        for (int i = 0; i < 8; ++i) acc[t][i] = 0.0f;
}

// ISA layouts (CDNA5 ISA):
//   A: lane<16 -> row=lane,    elems 0..7 = K0..7,  8..15 = K16..23
//      lane>=16-> row=lane-16, elems 0..7 = K8..15, 8..15 = K24..31
//   B: col = lane&15, elem i -> K = (lane<16 ? 0 : 8) + i + (i>=8 ? 8 : 0), same as A
//   C/D: elem r -> row = r + (lane>=16 ? 8 : 0), col = lane&15
//
// MT row-tiles of 16 share each B fragment. acc layout: acc[mt*NTD + nt].
template<int KSTEPS, int NTT, int NTD, int MT>
__device__ __forceinline__ void wmma_gemm(const bf16_t* a_lds, int astride,
                                          const bf16_t* Bp, int nt0,
                                          v8f* acc, int lane)
{
    const int lo = (lane & 16) ? 8 : 0;
    const int arowoff = (lane & 15) * astride;
    for (int ks = 0; ks < KSTEPS; ++ks) {
        v16bf a[MT];
#pragma unroll
        for (int mt = 0; mt < MT; ++mt) {
            const bf16_t* arow = a_lds + (mt * 16) * astride + arowoff;
            v8bf a0 = *(const v8bf*)(arow + ks * 32 + lo);
            v8bf a1 = *(const v8bf*)(arow + ks * 32 + lo + 16);
#pragma unroll
            for (int i = 0; i < 8; ++i) { a[mt][i] = a0[i]; a[mt][i + 8] = a1[i]; }
        }
#pragma unroll
        for (int nt = 0; nt < NTD; ++nt) {
            v16bf b = *(const v16bf*)(Bp + (size_t)((ks * NTT + nt0 + nt) * 32 + lane) * 16);
#pragma unroll
            for (int mt = 0; mt < MT; ++mt)
                acc[mt * NTD + nt] = __builtin_amdgcn_wmma_f32_16x16x32_bf16(
                    false, a[mt], false, b, (short)0, acc[mt * NTD + nt], false, false);
        }
    }
}

// bias + ReLU, write (MT*16) x (16*NTD) tile to LDS as bf16 (A for next GEMM)
template<int NTD, int MT>
__device__ __forceinline__ void bias_relu_to_lds(const v8f* acc, const float* bias,
                                                 int ncol0, bf16_t* h, int hstride,
                                                 int lane)
{
    const int col = lane & 15;
    const int rowoff = (lane & 16) ? 8 : 0;
#pragma unroll
    for (int mt = 0; mt < MT; ++mt)
#pragma unroll
        for (int nt = 0; nt < NTD; ++nt) {
            float bv = bias[ncol0 + nt * 16 + col];
#pragma unroll
            for (int r = 0; r < 8; ++r) {
                float v = acc[mt * NTD + nt][r] + bv;
                v = v > 0.0f ? v : 0.0f;
                h[(mt * 16 + r + rowoff) * hstride + ncol0 + nt * 16 + col] = f2bf(v);
            }
        }
}

// ---------------------------------------------------------------------------
// Pack row-major f32 weight [K, Nout] into per-lane WMMA B fragments (bf16),
// K zero-padded to Kpad (multiple of 32). Fragment order: [ks][nt][lane][16].
// ---------------------------------------------------------------------------
__global__ void pack_weight_kernel(const float* __restrict__ W, bf16_t* __restrict__ out,
                                   int K, int Kpad, int Nout)
{
    int idx = blockIdx.x * blockDim.x + threadIdx.x;
    int total = (Kpad / 32) * (Nout / 16) * 32 * 16;
    if (idx >= total) return;
    int i    = idx & 15;
    int lane = (idx >> 4) & 31;
    int frag = idx >> 9;
    int NT   = Nout >> 4;
    int nt   = frag % NT;
    int ks   = frag / NT;
    int k = ks * 32 + ((lane & 16) ? 8 : 0) + i + ((i >= 8) ? 8 : 0);
    int n = nt * 16 + (lane & 15);
    float v = (k < K) ? W[(size_t)k * Nout + n] : 0.0f;
    out[idx] = f2bf(v);
}

__global__ void zero_kernel(float* p, int n)
{
    int i = blockIdx.x * blockDim.x + threadIdx.x;
    if (i < n) p[i] = 0.0f;
}

// ---------------------------------------------------------------------------
// Edge kernel: one wave = 32 edges (two 16-row A tiles). Build
// feat=[xi, xj-xi, xj*xi, |d|, <xj,xi>, edge_attr, 0-pad] in LDS (bf16),
// MLP (EIN->128->128) via WMMA, then segment-max scatter with int-bit
// atomicMax. 2 waves / block; E divisible by 64.
// ---------------------------------------------------------------------------
template<int F, int KPAD>
__global__ void __launch_bounds__(64) edge_conv_kernel(
    const float* __restrict__ x, const int* __restrict__ srcI, const int* __restrict__ dstI,
    const float* __restrict__ eattr,
    const bf16_t* __restrict__ W1p, const float* __restrict__ b1,
    const bf16_t* __restrict__ W2p, const float* __restrict__ b2,
    float* __restrict__ agg)
{
    extern __shared__ bf16_t smem[];
    const int lane = threadIdx.x & 31;
    const int wave = threadIdx.x >> 5;
    bf16_t* feat = smem + (size_t)wave * 32 * KPAD;
    const int ebase = (blockIdx.x * 2 + wave) * 32;

    for (int r = 0; r < 32; ++r) {
        const int e  = ebase + r;
        const int sj = srcI[e];
        const int di = dstI[e];
        bf16_t* fr = feat + r * KPAD;
        float sum_d2 = 0.0f, sum_p = 0.0f;
        if constexpr (F == 32) {
            float xi = x[(size_t)di * 32 + lane];
            float xj = x[(size_t)sj * 32 + lane];
            float d = xj - xi, p = xj * xi;
            fr[lane]      = f2bf(xi);
            fr[32 + lane] = f2bf(d);
            fr[64 + lane] = f2bf(p);
            sum_d2 = d * d; sum_p = p;
        } else {
            const float4 xi4 = ((const float4*)(x + (size_t)di * 128))[lane];
            const float4 xj4 = ((const float4*)(x + (size_t)sj * 128))[lane];
            float xi[4] = {xi4.x, xi4.y, xi4.z, xi4.w};
            float xj[4] = {xj4.x, xj4.y, xj4.z, xj4.w};
#pragma unroll
            for (int j = 0; j < 4; ++j) {
                float d = xj[j] - xi[j], p = xj[j] * xi[j];
                int c = lane * 4 + j;
                fr[c]       = f2bf(xi[j]);
                fr[128 + c] = f2bf(d);
                fr[256 + c] = f2bf(p);
                sum_d2 += d * d; sum_p += p;
            }
        }
#pragma unroll
        for (int off = 16; off > 0; off >>= 1) {
            sum_d2 += __shfl_xor(sum_d2, off, 32);
            sum_p  += __shfl_xor(sum_p,  off, 32);
        }
        const int b = 3 * F;
        if (lane == 0) { fr[b] = f2bf(sqrtf(sum_d2)); fr[b + 1] = f2bf(sum_p); }
        if (lane < 4)  fr[b + 2 + lane] = f2bf(eattr[(size_t)e * 4 + lane]);
        for (int k = b + 6 + lane; k < KPAD; k += 32)
            fr[k] = __builtin_bit_cast(bf16_t, (unsigned short)0);
    }

    v8f acc[16];
    zero_acc<16>(acc);
    wmma_gemm<KPAD / 32, 8, 8, 2>(feat, KPAD, W1p, 0, acc, lane);

    bf16_t* h = feat;                       // reuse LDS region (GEMM1 A fully consumed)
    bias_relu_to_lds<8, 2>(acc, b1, 0, h, 128, lane);

    v8f acc2[16];
    zero_acc<16>(acc2);
    wmma_gemm<4, 8, 8, 2>(h, 128, W2p, 0, acc2, lane);

    const int col = lane & 15;
    const int rowoff = (lane & 16) ? 8 : 0;
#pragma unroll
    for (int mt = 0; mt < 2; ++mt)
#pragma unroll
        for (int nt = 0; nt < 8; ++nt) {
            float bv = b2[nt * 16 + col];
#pragma unroll
            for (int r = 0; r < 8; ++r) {
                float v = acc2[mt * 8 + nt][r] + bv;
                v = v > 0.0f ? v : 0.0f;
                int e  = ebase + mt * 16 + r + rowoff;
                int dn = dstI[e];
                atomicMax((int*)(agg + (size_t)dn * 128 + nt * 16 + col), __float_as_int(v));
            }
        }
}

// ---------------------------------------------------------------------------
// Node kernel: y = relu(relu([x, agg] @ lW1 + lb1) @ lW2 + lb2) (+x residual)
// One wave = 32 nodes; 2 waves / block.
// ---------------------------------------------------------------------------
template<int F, bool RES>
__global__ void __launch_bounds__(64) node_mlp_kernel(
    const float* __restrict__ x, const float* __restrict__ agg,
    const bf16_t* __restrict__ W1p, const float* __restrict__ b1,
    const bf16_t* __restrict__ W2p, const float* __restrict__ b2,
    float* __restrict__ y, int Ncount)
{
    constexpr int K1 = F + 128;
    extern __shared__ bf16_t smem[];
    const int lane = threadIdx.x & 31;
    const int wave = threadIdx.x >> 5;
    bf16_t* st = smem + (size_t)wave * 32 * K1;
    const int nbase = (blockIdx.x * 2 + wave) * 32;

    for (int r = 0; r < 32; ++r) {
        int n  = nbase + r;
        int nc = n < Ncount ? n : 0;
        bf16_t* sr = st + r * K1;
        for (int k = lane; k < F; k += 32)   sr[k]     = f2bf(x[(size_t)nc * F + k]);
        for (int k = lane; k < 128; k += 32) sr[F + k] = f2bf(agg[(size_t)nc * 128 + k]);
    }

    v8f acc[16];
    zero_acc<16>(acc);
    wmma_gemm<K1 / 32, 8, 8, 2>(st, K1, W1p, 0, acc, lane);
    bias_relu_to_lds<8, 2>(acc, b1, 0, st, 128, lane);

    v8f acc2[16];
    zero_acc<16>(acc2);
    wmma_gemm<4, 8, 8, 2>(st, 128, W2p, 0, acc2, lane);

    const int col = lane & 15;
    const int rowoff = (lane & 16) ? 8 : 0;
#pragma unroll
    for (int mt = 0; mt < 2; ++mt)
#pragma unroll
        for (int nt = 0; nt < 8; ++nt) {
            float bv = b2[nt * 16 + col];
#pragma unroll
            for (int r = 0; r < 8; ++r) {
                int n = nbase + mt * 16 + r + rowoff;
                if (n < Ncount) {
                    float v = acc2[mt * 8 + nt][r] + bv;
                    v = v > 0.0f ? v : 0.0f;
                    if (RES) v += x[(size_t)n * F + nt * 16 + col];
                    y[(size_t)n * 128 + nt * 16 + col] = v;
                }
            }
        }
}

// ---------------------------------------------------------------------------
// Fusion: out = relu(relu([x1,x2,x3] @ W1 + b1) @ W2 + b2), 384 -> 256 -> 128
// One wave = 32 nodes; 1 wave / block; hidden 256 done as two 8-tile chunks.
// ---------------------------------------------------------------------------
__global__ void __launch_bounds__(32) fusion_kernel(
    const float* __restrict__ x1, const float* __restrict__ x2, const float* __restrict__ x3,
    const bf16_t* __restrict__ W1p, const float* __restrict__ b1,
    const bf16_t* __restrict__ W2p, const float* __restrict__ b2,
    float* __restrict__ out, int Ncount)
{
    constexpr int K1 = 384, H = 256;
    extern __shared__ bf16_t smem[];
    const int lane = threadIdx.x & 31;
    bf16_t* st = smem;
    bf16_t* h  = st + 32 * K1;
    const int nbase = blockIdx.x * 32;

    for (int r = 0; r < 32; ++r) {
        int n  = nbase + r;
        int nc = n < Ncount ? n : 0;
        bf16_t* sr = st + r * K1;
        for (int k = lane; k < 128; k += 32) {
            sr[k]       = f2bf(x1[(size_t)nc * 128 + k]);
            sr[128 + k] = f2bf(x2[(size_t)nc * 128 + k]);
            sr[256 + k] = f2bf(x3[(size_t)nc * 128 + k]);
        }
    }

    for (int chunk = 0; chunk < 2; ++chunk) {
        v8f acc[16];
        zero_acc<16>(acc);
        wmma_gemm<K1 / 32, 16, 8, 2>(st, K1, W1p, chunk * 8, acc, lane);
        bias_relu_to_lds<8, 2>(acc, b1, chunk * 128, h, H, lane);
    }

    v8f acc2[16];
    zero_acc<16>(acc2);
    wmma_gemm<H / 32, 8, 8, 2>(h, H, W2p, 0, acc2, lane);

    const int col = lane & 15;
    const int rowoff = (lane & 16) ? 8 : 0;
#pragma unroll
    for (int mt = 0; mt < 2; ++mt)
#pragma unroll
        for (int nt = 0; nt < 8; ++nt) {
            float bv = b2[nt * 16 + col];
#pragma unroll
            for (int r = 0; r < 8; ++r) {
                int n = nbase + mt * 16 + r + rowoff;
                if (n < Ncount) {
                    float v = acc2[mt * 8 + nt][r] + bv;
                    v = v > 0.0f ? v : 0.0f;
                    out[(size_t)n * 128 + nt * 16 + col] = v;
                }
            }
        }
}

// ---------------------------------------------------------------------------
extern "C" void kernel_launch(void* const* d_in, const int* in_sizes, int n_in,
                              void* d_out, int out_size, void* d_ws, size_t ws_size,
                              hipStream_t stream)
{
    (void)in_sizes; (void)n_in; (void)out_size; (void)ws_size;
    const int N = 25000, E = 400000;

    const float* x    = (const float*)d_in[0];
    const int*   eidx = (const int*)d_in[1];
    const float* ea   = (const float*)d_in[2];
    const int* srcI = eidx;
    const int* dstI = eidx + E;

    const float* W[31];
    for (int i = 3; i < 31; ++i) W[i] = (const float*)d_in[i];

    // workspace carve
    char* ws = (char*)d_ws;
    size_t off = 0;
    auto alloc = [&](size_t bytes) -> void* {
        void* p = ws + off;
        off += (bytes + 255) & ~(size_t)255;
        return p;
    };
    float* x1  = (float*)alloc((size_t)N * 128 * 4);
    float* x2  = (float*)alloc((size_t)N * 128 * 4);
    float* x3  = (float*)alloc((size_t)N * 128 * 4);
    float* agg = (float*)alloc((size_t)N * 128 * 4);

    auto packed_bytes = [](int Kpad, int Nout) -> size_t {
        return (size_t)(Kpad / 32) * (Nout / 16) * 512 * 2;
    };
    bf16_t* c_e1p[3]; bf16_t* c_e2p[3]; bf16_t* c_l1p[3]; bf16_t* c_l2p[3];
    const int e_kpad[3] = {128, 416, 416};
    const int l_k1[3]   = {160, 256, 256};
    for (int l = 0; l < 3; ++l) {
        c_e1p[l] = (bf16_t*)alloc(packed_bytes(e_kpad[l], 128));
        c_e2p[l] = (bf16_t*)alloc(packed_bytes(128, 128));
        c_l1p[l] = (bf16_t*)alloc(packed_bytes(l_k1[l], 128));
        c_l2p[l] = (bf16_t*)alloc(packed_bytes(128, 128));
    }
    bf16_t* fusW1p = (bf16_t*)alloc(packed_bytes(384, 256));
    bf16_t* fusW2p = (bf16_t*)alloc(packed_bytes(256, 128));

    auto pack = [&](const float* Wp, bf16_t* o, int K, int Kpad, int Nout) {
        int total = (Kpad / 32) * (Nout / 16) * 512;
        pack_weight_kernel<<<(total + 255) / 256, 256, 0, stream>>>(Wp, o, K, Kpad, Nout);
    };
    const int e_in[3] = {102, 390, 390};
    for (int l = 0; l < 3; ++l) {
        int b = 3 + l * 8;
        pack(W[b + 0], c_e1p[l], e_in[l], e_kpad[l], 128);  // eW1
        pack(W[b + 2], c_e2p[l], 128, 128, 128);            // eW2
        pack(W[b + 4], c_l1p[l], l_k1[l], l_k1[l], 128);    // lW1
        pack(W[b + 6], c_l2p[l], 128, 128, 128);            // lW2
    }
    pack(W[27], fusW1p, 384, 384, 256);
    pack(W[29], fusW2p, 256, 256, 128);

    const int eblocks  = E / 64;                 // 2 waves * 32 edges per block
    const int ntiles32 = (N + 31) / 32;          // 782
    const int nblocks  = (ntiles32 + 1) / 2;     // 391
    const int zgrid    = (N * 128 + 255) / 256;

    // ----- layer 1 (F=32, e_in=102 -> pad 128), no residual -----
    zero_kernel<<<zgrid, 256, 0, stream>>>(agg, N * 128);
    edge_conv_kernel<32, 128><<<eblocks, 64, 2 * 32 * 128 * 2, stream>>>(
        x, srcI, dstI, ea, c_e1p[0], W[4], c_e2p[0], W[6], agg);
    node_mlp_kernel<32, false><<<nblocks, 64, 2 * 32 * 160 * 2, stream>>>(
        x, agg, c_l1p[0], W[8], c_l2p[0], W[10], x1, N);

    // ----- layer 2 (F=128, e_in=390 -> pad 416), residual -----
    zero_kernel<<<zgrid, 256, 0, stream>>>(agg, N * 128);
    edge_conv_kernel<128, 416><<<eblocks, 64, 2 * 32 * 416 * 2, stream>>>(
        x1, srcI, dstI, ea, c_e1p[1], W[12], c_e2p[1], W[14], agg);
    node_mlp_kernel<128, true><<<nblocks, 64, 2 * 32 * 256 * 2, stream>>>(
        x1, agg, c_l1p[1], W[16], c_l2p[1], W[18], x2, N);

    // ----- layer 3 -----
    zero_kernel<<<zgrid, 256, 0, stream>>>(agg, N * 128);
    edge_conv_kernel<128, 416><<<eblocks, 64, 2 * 32 * 416 * 2, stream>>>(
        x2, srcI, dstI, ea, c_e1p[2], W[20], c_e2p[2], W[22], agg);
    node_mlp_kernel<128, true><<<nblocks, 64, 2 * 32 * 256 * 2, stream>>>(
        x2, agg, c_l1p[2], W[24], c_l2p[2], W[26], x3, N);

    // ----- fusion -----
    fusion_kernel<<<ntiles32, 32, (32 * 384 + 32 * 256) * 2, stream>>>(
        x1, x2, x3, fusW1p, W[28], fusW2p, W[30], (float*)d_out, N);
}
